// HEATEncoder_55868934586451
// MI455X (gfx1250) — matrix operand order repaired and checked by reading, not compiled, not run on hardware
//
#include <hip/hip_runtime.h>

typedef __attribute__((ext_vector_type(16))) _Float16 v16h;
typedef __attribute__((ext_vector_type(8)))  float    v8f;

#define N_GEN 50000
#define N_BUS 80000
#define N_RES 10000
#define N_TOT 140000
#define HID   128
#define E_P   100000
#define E_S   100000
#define E_T   150000
#define E_B   50000
#define E_TOT 400000
#define NEGS  0.2f

__device__ __forceinline__ float lrelu(float x) { return x >= 0.f ? x : NEGS * x; }

// order-preserving float<->uint encoding for atomicMax-based segment max
__device__ __forceinline__ unsigned encf(float f) {
    unsigned u = __float_as_uint(f);
    return (u & 0x80000000u) ? ~u : (u | 0x80000000u);
}
__device__ __forceinline__ float decf(unsigned u) {
    unsigned v = (u & 0x80000000u) ? (u & 0x7fffffffu) : ~u;
    return __uint_as_float(v);
}

// Build 16x32 f16 A-fragment from an f32 row (ISA 16-bit A layout:
// lane<16: elems 0..7 = K0..7, 8..15 = K16..23; lane>=16: +8)
__device__ __forceinline__ v16h load_a_f32(const float* __restrict__ xr, int k0) {
    float4 f0 = *(const float4*)(xr + k0);
    float4 f1 = *(const float4*)(xr + k0 + 4);
    float4 f2 = *(const float4*)(xr + k0 + 16);
    float4 f3 = *(const float4*)(xr + k0 + 20);
    v16h a;
    a[0]=(_Float16)f0.x; a[1]=(_Float16)f0.y; a[2]=(_Float16)f0.z; a[3]=(_Float16)f0.w;
    a[4]=(_Float16)f1.x; a[5]=(_Float16)f1.y; a[6]=(_Float16)f1.z; a[7]=(_Float16)f1.w;
    a[8]=(_Float16)f2.x; a[9]=(_Float16)f2.y; a[10]=(_Float16)f2.z; a[11]=(_Float16)f2.w;
    a[12]=(_Float16)f3.x; a[13]=(_Float16)f3.y; a[14]=(_Float16)f3.z; a[15]=(_Float16)f3.w;
    return a;
}

// ---------------- input projection: one block (128 thr) per node ----------------
__global__ void k_proj(const float* __restrict__ xin, const float* __restrict__ W,
                       const float* __restrict__ b, float* __restrict__ xout,
                       int nNodes, int inDim, int rowBase) {
    __shared__ float sx[64];
    int node = blockIdx.x;
    if (node >= nNodes) return;
    int tid = threadIdx.x;  // 0..127 = output column
    if (tid < inDim) sx[tid] = xin[(size_t)node * inDim + tid];
    __syncthreads();
    float acc = b[tid];
    for (int k = 0; k < inDim; ++k) acc += sx[k] * W[k * HID + tid];
    xout[((size_t)(rowBase + node)) * HID + tid] = acc;
}

// ---------------- concat edge lists / attrs once ----------------
__global__ void k_edgeprep(const int* eiP, const int* eiS, const int* eiT, const int* eiB,
                           const float* eaP, const float* eaS, const float* eaT, const float* eaB,
                           int* __restrict__ srcA, int* __restrict__ dstA, float* __restrict__ eaC) {
    int e = blockIdx.x * blockDim.x + threadIdx.x;
    if (e >= E_TOT) return;
    const int* ei; const float* ea; int off, ne;
    if (e < E_P)              { ei = eiP; ea = eaP; off = e;                 ne = E_P; }
    else if (e < E_P+E_S)     { ei = eiS; ea = eaS; off = e - E_P;          ne = E_S; }
    else if (e < E_P+E_S+E_T) { ei = eiT; ea = eaT; off = e - (E_P+E_S);    ne = E_T; }
    else                      { ei = eiB; ea = eaB; off = e - (E_P+E_S+E_T); ne = E_B; }
    srcA[e] = ei[off];
    dstA[e] = ei[ne + off];
    eaC[e * 2 + 0] = ea[off * 2 + 0];
    eaC[e * 2 + 1] = ea[off * 2 + 1];
}

// ---------------- per-layer weight prep: f16 + transpose ----------------
// Wt[t][o][d] = (f16) hW[t][d][o]   (so B-frag lanes read contiguous K)
// Lt[o][k]    = (f16) linw[k][o], K padded 144->160 with zeros
__global__ void k_prepw(const float* __restrict__ hW, const float* __restrict__ linw,
                        _Float16* __restrict__ Wt, _Float16* __restrict__ Lt) {
    int i = blockIdx.x * blockDim.x + threadIdx.x;
    if (i < 3 * HID * HID) {
        int t = i >> 14, rem = i & 16383, o = rem >> 7, d = rem & 127;
        Wt[i] = (_Float16)hW[(t * HID + d) * HID + o];
    }
    if (i < HID * 160) {
        int o = i / 160, k = i % 160;
        Lt[i] = (k < 144) ? (_Float16)linw[k * HID + o] : (_Float16)0.0f;
    }
}

// ---------------- zero agg / softmax state ----------------
__global__ void k_zero(float* __restrict__ agg, unsigned* __restrict__ menc,
                       float* __restrict__ denom) {
    size_t i = (size_t)blockIdx.x * blockDim.x + threadIdx.x;
    if (i < (size_t)N_TOT * HID) agg[i] = 0.f;
    if (i < (size_t)N_TOT * 4) { menc[i] = 0u; denom[i] = 0.f; }
}

// ---------------- HeteroLinear via WMMA: one wave per 16-row tile ----------------
__global__ void __launch_bounds__(256)
k_hl(const float* __restrict__ x, const _Float16* __restrict__ Wt,
     const float* __restrict__ bias, float* __restrict__ y,
     int rowBase, int nTiles) {
    int gtid  = blockIdx.x * blockDim.x + threadIdx.x;
    int wave  = gtid >> 5;
    int lane  = threadIdx.x & 31;
    int nWav  = (gridDim.x * blockDim.x) >> 5;
    int m = lane & 15, hi = lane >> 4;
    for (int t = wave; t < nTiles; t += nWav) {
        const float* xr = x + (size_t)(rowBase + t * 16 + m) * HID;
        v16h A[4];
#pragma unroll
        for (int c = 0; c < 4; ++c) A[c] = load_a_f32(xr, c * 32 + hi * 8);
        // column-tile loop kept rolled: avoids hoisting 32 B-fragments -> scratch spill
#pragma unroll 1
        for (int nt = 0; nt < 8; ++nt) {
            int col = nt * 16 + m;  // B/D column of this lane
            v8f acc = {};
            const _Float16* bp = Wt + (size_t)col * HID + hi * 8;
#pragma unroll
            for (int c = 0; c < 4; ++c) {
                v16h b;
#pragma unroll
                for (int i = 0; i < 8; ++i) { b[i] = bp[c * 32 + i]; b[8 + i] = bp[c * 32 + 16 + i]; }
                acc = __builtin_amdgcn_wmma_f32_16x16x32_f16(
                          false, A[c], false, b, (short)0, acc, false, false);
            }
            float bv = bias[col];
            float* yo = y + (size_t)(rowBase + t * 16) * HID + col;
#pragma unroll
            for (int r = 0; r < 8; ++r)
                yo[(size_t)(r + hi * 8) * HID] = acc[r] + bv;
        }
    }
}

// ---------------- attention logits + segment-max, one thread per edge ----------------
__global__ void k_att(const float* __restrict__ y, const int* __restrict__ srcA,
                      const int* __restrict__ dstA, const float* __restrict__ eaC,
                      const float* __restrict__ ete, const float* __restrict__ eae,
                      const float* __restrict__ attw, float* __restrict__ logits,
                      unsigned* __restrict__ menc, _Float16* __restrict__ eaeeH) {
    int e = blockIdx.x * blockDim.x + threadIdx.x;
    if (e >= E_TOT) return;
    int et = (e < E_P) ? 0 : (e < E_P+E_S) ? 1 : (e < E_P+E_S+E_T) ? 2 : 3;
    int s = srcA[e], d = dstA[e];
    float ea0 = eaC[e * 2], ea1 = eaC[e * 2 + 1];
    float eaee[16], etee[16];
#pragma unroll
    for (int j = 0; j < 16; ++j) {
        eaee[j] = lrelu(ea0 * eae[j] + ea1 * eae[16 + j]);
        etee[j] = lrelu(ete[et * 16 + j]);
        eaeeH[(size_t)e * 16 + j] = (_Float16)eaee[j];
    }
    float a0 = 0.f, a1 = 0.f, a2 = 0.f, a3 = 0.f;
    const float* yi = y + (size_t)d * HID;
    const float* yj = y + (size_t)s * HID;
    for (int k4 = 0; k4 < 32; ++k4) {
        float4 v = *(const float4*)(yi + k4 * 4);
        const float* w = attw + k4 * 16;
        a0 += v.x*w[0] + v.y*w[4] + v.z*w[8]  + v.w*w[12];
        a1 += v.x*w[1] + v.y*w[5] + v.z*w[9]  + v.w*w[13];
        a2 += v.x*w[2] + v.y*w[6] + v.z*w[10] + v.w*w[14];
        a3 += v.x*w[3] + v.y*w[7] + v.z*w[11] + v.w*w[15];
    }
    for (int k4 = 0; k4 < 32; ++k4) {
        float4 v = *(const float4*)(yj + k4 * 4);
        const float* w = attw + (HID + k4 * 4) * 4;
        a0 += v.x*w[0] + v.y*w[4] + v.z*w[8]  + v.w*w[12];
        a1 += v.x*w[1] + v.y*w[5] + v.z*w[9]  + v.w*w[13];
        a2 += v.x*w[2] + v.y*w[6] + v.z*w[10] + v.w*w[14];
        a3 += v.x*w[3] + v.y*w[7] + v.z*w[11] + v.w*w[15];
    }
#pragma unroll
    for (int j = 0; j < 16; ++j) {
        const float* w  = attw + (256 + j) * 4;
        const float* w2 = attw + (272 + j) * 4;
        a0 += etee[j]*w[0] + eaee[j]*w2[0];
        a1 += etee[j]*w[1] + eaee[j]*w2[1];
        a2 += etee[j]*w[2] + eaee[j]*w2[2];
        a3 += etee[j]*w[3] + eaee[j]*w2[3];
    }
    a0 = lrelu(a0); a1 = lrelu(a1); a2 = lrelu(a2); a3 = lrelu(a3);
    logits[(size_t)e * 4 + 0] = a0; logits[(size_t)e * 4 + 1] = a1;
    logits[(size_t)e * 4 + 2] = a2; logits[(size_t)e * 4 + 3] = a3;
    atomicMax(&menc[d * 4 + 0], encf(a0));
    atomicMax(&menc[d * 4 + 1], encf(a1));
    atomicMax(&menc[d * 4 + 2], encf(a2));
    atomicMax(&menc[d * 4 + 3], encf(a3));
}

// ---------------- exp + denominator ----------------
__global__ void k_exp(const int* __restrict__ dstA, float* __restrict__ logits,
                      const unsigned* __restrict__ menc, float* __restrict__ denom) {
    int e = blockIdx.x * blockDim.x + threadIdx.x;
    if (e >= E_TOT) return;
    int d = dstA[e];
#pragma unroll
    for (int h = 0; h < 4; ++h) {
        float mv = decf(menc[d * 4 + h]);
        float p  = __expf(logits[(size_t)e * 4 + h] - mv);
        logits[(size_t)e * 4 + h] = p;
        atomicAdd(&denom[d * 4 + h], p);
    }
}

// ---------------- message GEMM + alpha-weighted scatter: one wave per 16-edge tile ----
__global__ void __launch_bounds__(256)
k_msg(const float* __restrict__ y, const int* __restrict__ srcA,
      const int* __restrict__ dstA, const float* __restrict__ p4,
      const float* __restrict__ denom, const _Float16* __restrict__ eaeeH,
      const _Float16* __restrict__ Lt, float* __restrict__ agg) {
    int gtid = blockIdx.x * blockDim.x + threadIdx.x;
    int wave = gtid >> 5;
    int lane = threadIdx.x & 31;
    int nWav = (gridDim.x * blockDim.x) >> 5;
    int m = lane & 15, hi = lane >> 4;
    const int nTiles = E_TOT / 16;
    for (int t = wave; t < nTiles; t += nWav) {
        int e = t * 16 + m;           // both half-waves compute row m's edge
        int s = srcA[e];
        int d = dstA[e];
        float4 pv = *(const float4*)(p4 + (size_t)e * 4);
        float4 dv = *(const float4*)(denom + (size_t)d * 4);
        float alpha = 0.25f * (pv.x / (dv.x + 1e-16f) + pv.y / (dv.y + 1e-16f) +
                               pv.z / (dv.z + 1e-16f) + pv.w / (dv.w + 1e-16f));
        // A fragments: K 0..127 from gathered node row, K 128..143 eaee, 144..159 zero
        const float* xr = y + (size_t)s * HID;
        v16h A[5];
#pragma unroll
        for (int c = 0; c < 4; ++c) A[c] = load_a_f32(xr, c * 32 + hi * 8);
        {
            v16h a;
            const _Float16* ep = eaeeH + (size_t)e * 16 + hi * 8;
#pragma unroll
            for (int i = 0; i < 8; ++i) a[i] = ep[i];
#pragma unroll
            for (int i = 8; i < 16; ++i) a[i] = (_Float16)0.0f;
            A[4] = a;
        }
        // keep rolled: avoids hoisting 40 B-fragments -> scratch spill
#pragma unroll 1
        for (int nt = 0; nt < 8; ++nt) {
            v8f acc = {};
            const _Float16* bp = Lt + (size_t)(nt * 16 + m) * 160 + hi * 8;
#pragma unroll
            for (int c = 0; c < 5; ++c) {
                v16h b;
#pragma unroll
                for (int i = 0; i < 8; ++i) { b[i] = bp[c * 32 + i]; b[8 + i] = bp[c * 32 + 16 + i]; }
                acc = __builtin_amdgcn_wmma_f32_16x16x32_f16(
                          false, A[c], false, b, (short)0, acc, false, false);
            }
#pragma unroll
            for (int r = 0; r < 8; ++r) {
                int row = r + hi * 8;
                int   dr = __shfl(d, row, 32);
                float ar = __shfl(alpha, row, 32);
                atomicAdd(&agg[(size_t)dr * HID + nt * 16 + m], acc[r] * ar);
            }
        }
    }
}

// ---------------- relu + residual ----------------
__global__ void k_post(const float* __restrict__ agg, const float* __restrict__ resid,
                       float* __restrict__ xout, int hasResid) {
    size_t i = (size_t)blockIdx.x * blockDim.x + threadIdx.x;
    if (i >= (size_t)N_TOT * HID) return;
    float v = agg[i];
    v = v > 0.f ? v : 0.f;
    if (hasResid) v += resid[i];
    xout[i] = v;
}

extern "C" void kernel_launch(void* const* d_in, const int* in_sizes, int n_in,
                              void* d_out, int out_size, void* d_ws, size_t ws_size,
                              hipStream_t stream) {
    const float* x_gen = (const float*)d_in[0];
    const float* x_bus = (const float*)d_in[1];
    const float* x_res = (const float*)d_in[2];
    const int*   eiP   = (const int*)d_in[3];
    const int*   eiS   = (const int*)d_in[4];
    const int*   eiT   = (const int*)d_in[5];
    const int*   eiB   = (const int*)d_in[6];
    const float* eaP   = (const float*)d_in[7];
    const float* eaS   = (const float*)d_in[8];
    const float* eaT   = (const float*)d_in[9];
    const float* eaB   = (const float*)d_in[10];
    const float* pW_gen = (const float*)d_in[11]; const float* pb_gen = (const float*)d_in[12];
    const float* pW_bus = (const float*)d_in[13]; const float* pb_bus = (const float*)d_in[14];
    const float* pW_res = (const float*)d_in[15]; const float* pb_res = (const float*)d_in[16];
    const float* hWl[3]   = {(const float*)d_in[17], (const float*)d_in[23], (const float*)d_in[29]};
    const float* hbl[3]   = {(const float*)d_in[18], (const float*)d_in[24], (const float*)d_in[30]};
    const float* etel[3]  = {(const float*)d_in[19], (const float*)d_in[25], (const float*)d_in[31]};
    const float* eael[3]  = {(const float*)d_in[20], (const float*)d_in[26], (const float*)d_in[32]};
    const float* attwl[3] = {(const float*)d_in[21], (const float*)d_in[27], (const float*)d_in[33]};
    const float* linwl[3] = {(const float*)d_in[22], (const float*)d_in[28], (const float*)d_in[34]};
    float* out = (float*)d_out;

    // workspace carve-up (256B aligned blocks)
    char* wsp = (char*)d_ws;
    size_t off = 0;
    auto carve = [&](size_t bytes) -> void* {
        void* p = wsp + off;
        off += (bytes + 255) & ~(size_t)255;
        return p;
    };
    float*     b0     = (float*)carve((size_t)N_TOT * HID * 4);
    float*     b1     = (float*)carve((size_t)N_TOT * HID * 4);
    float*     b2     = (float*)carve((size_t)N_TOT * HID * 4);   // agg
    int*       srcA   = (int*)carve((size_t)E_TOT * 4);
    int*       dstA   = (int*)carve((size_t)E_TOT * 4);
    float*     eaC    = (float*)carve((size_t)E_TOT * 2 * 4);
    float*     logits = (float*)carve((size_t)E_TOT * 4 * 4);
    unsigned*  menc   = (unsigned*)carve((size_t)N_TOT * 4 * 4);
    float*     denom  = (float*)carve((size_t)N_TOT * 4 * 4);
    _Float16*  eaeeH  = (_Float16*)carve((size_t)E_TOT * 16 * 2);
    _Float16*  Wt     = (_Float16*)carve((size_t)3 * HID * HID * 2);
    _Float16*  Lt     = (_Float16*)carve((size_t)HID * 160 * 2);

    // input projections -> b0
    k_proj<<<N_GEN, 128, 0, stream>>>(x_gen, pW_gen, pb_gen, b0, N_GEN, 39, 0);
    k_proj<<<N_BUS, 128, 0, stream>>>(x_bus, pW_bus, pb_bus, b0, N_BUS, 16, N_GEN);
    k_proj<<<N_RES, 128, 0, stream>>>(x_res, pW_res, pb_res, b0, N_RES, 16, N_GEN + N_BUS);

    // edge concat (once)
    k_edgeprep<<<(E_TOT + 255) / 256, 256, 0, stream>>>(eiP, eiS, eiT, eiB,
                                                        eaP, eaS, eaT, eaB,
                                                        srcA, dstA, eaC);

    float* xin  = b0;
    float* aggb = b2;
    const int zeroBlocks = (N_TOT * HID + 255) / 256;
    for (int l = 0; l < 3; ++l) {
        float* yb = (xin == b0) ? b1 : b0;

        k_prepw<<<(3 * HID * HID + 255) / 256, 256, 0, stream>>>(hWl[l], linwl[l], Wt, Lt);
        k_zero<<<zeroBlocks, 256, 0, stream>>>(aggb, menc, denom);

        // HeteroLinear (type boundaries are 16-aligned -> uniform-type tiles)
        k_hl<<<(N_GEN / 16 + 7) / 8, 256, 0, stream>>>(xin, Wt + 0 * HID * HID,
                                                       hbl[l] + 0 * HID, yb, 0, N_GEN / 16);
        k_hl<<<(N_BUS / 16 + 7) / 8, 256, 0, stream>>>(xin, Wt + 1 * HID * HID,
                                                       hbl[l] + 1 * HID, yb, N_GEN, N_BUS / 16);
        k_hl<<<(N_RES / 16 + 7) / 8, 256, 0, stream>>>(xin, Wt + 2 * HID * HID,
                                                       hbl[l] + 2 * HID, yb, N_GEN + N_BUS, N_RES / 16);

        k_att<<<(E_TOT + 127) / 128, 128, 0, stream>>>(yb, srcA, dstA, eaC,
                                                       etel[l], eael[l], attwl[l],
                                                       logits, menc, eaeeH);
        k_exp<<<(E_TOT + 255) / 256, 256, 0, stream>>>(dstA, logits, menc, denom);
        k_msg<<<(E_TOT / 16 + 7) / 8, 256, 0, stream>>>(yb, srcA, dstA, logits, denom,
                                                        eaeeH, Lt, aggb);

        float* xnext = (l == 2) ? out : yb;
        k_post<<<zeroBlocks, 256, 0, stream>>>(aggb, xin, xnext, l > 0);
        xin = xnext;
    }
}
